// HapbertaAxialEncoder_28836410426041
// MI455X (gfx1250) — matrix operand order repaired and checked by reading, not compiled, not run on hardware
//
#include <hip/hip_runtime.h>
#include <cmath>

#define LNUM 2
#define DIM  512
#define NH   8
#define HDIM 64
#define IDIM 2048
#define BB   2
#define RR   32
#define SS   512
#define NBK  64
#define NT   (BB*RR*SS)   /* 32768 tokens, token t = ((b*RR+r)*SS+s) */

typedef __attribute__((ext_vector_type(16))) __bf16       v16bf;
typedef __attribute__((ext_vector_type(8)))  float        v8f;
typedef __attribute__((ext_vector_type(4)))  unsigned int v4u;

union Frag { v16bf h; v4u q[2]; };

__device__ __forceinline__ __bf16 f2bf(float f) {
  unsigned u = __builtin_bit_cast(unsigned, f);
  unsigned r = u + 0x7FFFu + ((u >> 16) & 1u);       // round-to-nearest-even
  unsigned short hs = (unsigned short)(r >> 16);
  return __builtin_bit_cast(__bf16, hs);
}

// ---------------------------------------------------------------------------
// Generic WMMA GEMM:  C(Mrows x Ncols) = A(Mrows x K, bf16 row-major)
//                                       x Bt(Ncols x K, bf16 row-major)^T
// One wave computes a 32 x (NSUB*16) tile: 2 M-subtiles x NSUB N-subtiles,
// K stepped by 32 with v_wmma_f32_16x16x32_bf16.  All tile indices are made
// wave-uniform via readfirstlane so no EXEC-mask predication is generated.
// MODE selects the epilogue (bias/residual/gelu/permuted stores).
// ---------------------------------------------------------------------------
template<int MODE, int NSUB>
__global__ __launch_bounds__(128)
void gemm_k(const __bf16* __restrict__ A, const __bf16* __restrict__ Bt,
            const float* __restrict__ bias, void* __restrict__ outp,
            int Mrows, int Ncols, int K, int ldout, float scale,
            long long sA, long long sB, long long sO)
{
  const int z = blockIdx.y;
  A  += (long long)z * sA;
  Bt += (long long)z * sB;
  const int nt   = (Ncols + 63) >> 6;
  const int mt   = Mrows >> 5;                      // 32-row M tiles
  const int wv   = __builtin_amdgcn_readfirstlane(threadIdx.x >> 5);
  const int tile = blockIdx.x * 4 + wv;             // scalar
  if (tile >= mt * nt) return;
  const int mtile = tile / nt;                      // scalar
  const int ntg   = tile % nt;                      // scalar
  const int lane  = threadIdx.x & 31;
  const int l15   = lane & 15;
  const int lh    = lane >> 4;                      // half-wave select

  v8f acc[2][NSUB] = {};

  // A 16-bit fragment layout (ISA 7.12.2): lanes 0-15 hold K = kb..kb+7 and
  // kb+16..kb+23 with kb=0; lanes 16-31 the same with kb=8.
  const __bf16* Arow0 = A + (long long)(mtile * 32 + l15) * K + (lh << 3);
  const __bf16* Arow1 = Arow0 + (long long)16 * K;
  // B fragment: same K layout as A (lanes 0-15 K=k0..k0+7 and k0+16..k0+23,
  // lanes 16-31 the same shifted by 8).
  const __bf16* Brow[NSUB];
  #pragma unroll
  for (int t = 0; t < NSUB; ++t)
    Brow[t] = Bt + (long long)((ntg << 6) + (t << 4) + l15) * K + (lh << 3);

  for (int k0 = 0; k0 < K; k0 += 32) {
    Frag a0, a1, b[NSUB];
    a0.q[0] = *(const v4u*)(Arow0 + k0);
    a0.q[1] = *(const v4u*)(Arow0 + k0 + 16);
    a1.q[0] = *(const v4u*)(Arow1 + k0);
    a1.q[1] = *(const v4u*)(Arow1 + k0 + 16);
    #pragma unroll
    for (int t = 0; t < NSUB; ++t) {
      b[t].q[0] = *(const v4u*)(Brow[t] + k0);
      b[t].q[1] = *(const v4u*)(Brow[t] + k0 + 16);
    }
    #pragma unroll
    for (int t = 0; t < NSUB; ++t)
      acc[0][t] = __builtin_amdgcn_wmma_f32_16x16x32_bf16(
          false, a0.h, false, b[t].h, (short)0, acc[0][t], false, false);
    #pragma unroll
    for (int t = 0; t < NSUB; ++t)
      acc[1][t] = __builtin_amdgcn_wmma_f32_16x16x32_bf16(
          false, a1.h, false, b[t].h, (short)0, acc[1][t], false, false);
  }

  // Epilogue.  C/D layout: lane 0-15 vgpr v -> M=v, lane 16-31 -> M=8+v, N=l15.
  #pragma unroll
  for (int ms = 0; ms < 2; ++ms) {
    #pragma unroll
    for (int t = 0; t < NSUB; ++t) {
      const int n = (ntg << 6) + (t << 4) + l15;
      float bv = 0.f;
      if (MODE <= 6) bv = bias[n];
      #pragma unroll
      for (int v = 0; v < 8; ++v) {
        const int m = mtile * 32 + ms * 16 + (lh << 3) + v;
        float val = acc[ms][t][v] + bv;
        if (MODE == 0) {                       // residual accumulate into f32 X
          float* o = (float*)outp;
          long long idx = (long long)m * ldout + n;
          o[idx] = o[idx] + val;
        } else if (MODE == 1) {                // plain bf16 store
          ((__bf16*)outp)[(long long)m * ldout + n] = f2bf(val * scale);
        } else if (MODE == 2) {                // exact GELU -> bf16
          float g = 0.5f * val * (1.f + erff(val * 0.70710678118654752f));
          ((__bf16*)outp)[(long long)m * ldout + n] = f2bf(g);
        } else if (MODE >= 3 && MODE <= 6) {   // QKV permuted stores
          const int b  = m / (RR * SS);
          const int r  = (m / SS) % RR;
          const int s  = m % SS;
          const int h  = n >> 6;
          const int hd = n & 63;
          long long idx;
          if (MODE == 3)       // row q/k:   [(h,b), s, r*HD+hd]
            idx = ((long long)((h*BB + b)*SS + s)) * (RR*HDIM) + r*HDIM + hd;
          else if (MODE == 4)  // row v^T:   [(h,b), r, hd, s]
            idx = ((long long)((h*BB + b)*RR + r)) * (HDIM*SS) + (long long)hd*SS + s;
          else if (MODE == 5)  // col q/k:   [(h,s,b), r*HD+hd]
            idx = ((long long)((h*SS + s)*BB + b)) * (RR*HDIM) + r*HDIM + hd;
          else                 // col v^T:   [(h,s,b), hd, r]
            idx = ((long long)((h*SS + s)*BB + b)) * (HDIM*RR) + hd*RR + r;
          ((__bf16*)outp)[idx] = f2bf(val * scale);
        } else if (MODE == 7) {                // raw f32 scores
          ((float*)outp)[(long long)z * sO + (long long)m * ldout + n] = val;
        } else if (MODE == 8) {                // row ctx -> token-major bf16
          const int h = z / BB, b = z % BB;
          const int r = n >> 6, d = n & 63;
          long long tok = (long long)(b*RR + r) * SS + m;
          ((__bf16*)outp)[tok * DIM + h*HDIM + d] = f2bf(val);
        } else {                               // 9: col ctx -> token-major bf16
          const int b = z % BB;
          const int s = (z / BB) % SS;
          const int h = z / (BB * SS);
          long long tok = (long long)(b*RR + m) * SS + s;
          ((__bf16*)outp)[tok * DIM + h*HDIM + n] = f2bf(val);
        }
      }
    }
  }
}

// ---------------------------------------------------------------------------
// LayerNorm over D=512, one wave per token (wave32: 16 elems/lane).
// ---------------------------------------------------------------------------
template<bool OUTBF>
__global__ __launch_bounds__(256)
void ln_k(const float* __restrict__ x, const float* __restrict__ gam,
          const float* __restrict__ bet, void* __restrict__ out)
{
  const int t    = blockIdx.x * 8 + (threadIdx.x >> 5);
  const int lane = threadIdx.x & 31;
  const float* xr = x + (long long)t * DIM;
  float vals[16];
  float s = 0.f, s2 = 0.f;
  #pragma unroll
  for (int i = 0; i < 16; ++i) {
    float v = xr[lane + 32*i];
    vals[i] = v; s += v; s2 += v*v;
  }
  #pragma unroll
  for (int o = 16; o > 0; o >>= 1) { s += __shfl_xor(s, o, 32); s2 += __shfl_xor(s2, o, 32); }
  const float mean = s * (1.f/DIM);
  const float var  = fmaxf(s2 * (1.f/DIM) - mean*mean, 0.f);
  const float rstd = rsqrtf(var + 1e-12f);
  #pragma unroll
  for (int i = 0; i < 16; ++i) {
    const int c = lane + 32*i;
    float y = (vals[i] - mean) * rstd * gam[c] + bet[c];
    if (OUTBF) ((__bf16*)out)[(long long)t*DIM + c] = f2bf(y);
    else       ((float*)out)[(long long)t*DIM + c] = y;
  }
}

// ---------------------------------------------------------------------------
// Row-attn softmax over j (512), with mask + T5 distance-bucket bias.
// One wave per (h,b,i) row.  boundaries = linspace(0, MAXD, 63), step MAXD/62
// -> searchsorted_left(|d|) == min(63, ceil(|d| * 62/MAXD)).
// ---------------------------------------------------------------------------
__global__ __launch_bounds__(256)
void softmax_row_k(const float* __restrict__ SC, const float* __restrict__ mask,
                   const float* __restrict__ dist, const float* __restrict__ bemb,
                   __bf16* __restrict__ PB)
{
  const int rowid = blockIdx.x * 8 + (threadIdx.x >> 5);  // z*512 + i, z=(h,b)
  const int lane  = threadIdx.x & 31;
  const int i = rowid & 511;
  const int zz = rowid >> 9;
  const int h = zz >> 1, b = zz & 1;
  const float* sr = SC   + (long long)rowid * SS;
  const float* mr = mask + ((long long)b*SS + i)*SS;
  const float* dr = dist + ((long long)b*SS + i)*SS;
  float w[16];
  float mx = -3.402823466e38f;
  #pragma unroll
  for (int q = 0; q < 16; ++q) {
    const int j = lane + 32*q;
    float v = sr[j] + (1.0f - mr[j]) * (-3.402823466e38f);
    int idx = (int)ceilf(fabsf(dr[j]) * (62.0f / 50000.0f));
    idx = idx < 0 ? 0 : (idx > 63 ? 63 : idx);
    v += bemb[idx*NH + h];
    w[q] = v; mx = fmaxf(mx, v);
  }
  #pragma unroll
  for (int o = 16; o > 0; o >>= 1) mx = fmaxf(mx, __shfl_xor(mx, o, 32));
  float ssum = 0.f;
  #pragma unroll
  for (int q = 0; q < 16; ++q) { w[q] = __expf(w[q] - mx); ssum += w[q]; }
  #pragma unroll
  for (int o = 16; o > 0; o >>= 1) ssum += __shfl_xor(ssum, o, 32);
  const float inv = 1.f / ssum;
  __bf16* pr = PB + (long long)rowid * SS;
  #pragma unroll
  for (int q = 0; q < 16; ++q) pr[lane + 32*q] = f2bf(w[q] * inv);
}

// Column-attn softmax over j (32): one wave per row, one element per lane.
__global__ __launch_bounds__(256)
void softmax_col_k(const float* __restrict__ SC, __bf16* __restrict__ PB)
{
  const long long rowid = (long long)blockIdx.x * 8 + (threadIdx.x >> 5);
  const int lane = threadIdx.x & 31;
  float v = SC[rowid * RR + lane];
  float mx = v;
  #pragma unroll
  for (int o = 16; o > 0; o >>= 1) mx = fmaxf(mx, __shfl_xor(mx, o, 32));
  float e = __expf(v - mx), ssum = e;
  #pragma unroll
  for (int o = 16; o > 0; o >>= 1) ssum += __shfl_xor(ssum, o, 32);
  PB[rowid * RR + lane] = f2bf(e / ssum);
}

// Transpose + f32->bf16 weight conversion: W(Kd x Md) -> Wt(Md x Kd).
__global__ __launch_bounds__(256)
void wconv_k(const float* __restrict__ W, __bf16* __restrict__ Wt, int Kd, int Md)
{
  long long i = (long long)blockIdx.x * 256 + threadIdx.x;
  if (i >= (long long)Kd * Md) return;
  int m = (int)(i / Kd), k = (int)(i % Kd);
  Wt[i] = f2bf(W[(long long)k * Md + m]);
}

// ---------------------------------------------------------------------------
static inline void launch_gemm(int mode, const __bf16* A, const __bf16* Bt,
                               const float* bias, void* out,
                               int Mrows, int Ncols, int K, int ldout, float scale,
                               long long sA, long long sB, long long sO, int batch,
                               hipStream_t stream)
{
  const int tiles = (Mrows >> 5) * ((Ncols + 63) >> 6);
  dim3 grid((unsigned)((tiles + 3) / 4), (unsigned)batch);
  dim3 blk(128);
  const bool half_n = (Ncols & 63) != 0;   // only the 32-wide col-score GEMM
  switch (mode) {
    case 0: gemm_k<0,4><<<grid, blk, 0, stream>>>(A, Bt, bias, out, Mrows, Ncols, K, ldout, scale, sA, sB, sO); break;
    case 2: gemm_k<2,4><<<grid, blk, 0, stream>>>(A, Bt, bias, out, Mrows, Ncols, K, ldout, scale, sA, sB, sO); break;
    case 3: gemm_k<3,4><<<grid, blk, 0, stream>>>(A, Bt, bias, out, Mrows, Ncols, K, ldout, scale, sA, sB, sO); break;
    case 4: gemm_k<4,4><<<grid, blk, 0, stream>>>(A, Bt, bias, out, Mrows, Ncols, K, ldout, scale, sA, sB, sO); break;
    case 5: gemm_k<5,4><<<grid, blk, 0, stream>>>(A, Bt, bias, out, Mrows, Ncols, K, ldout, scale, sA, sB, sO); break;
    case 6: gemm_k<6,4><<<grid, blk, 0, stream>>>(A, Bt, bias, out, Mrows, Ncols, K, ldout, scale, sA, sB, sO); break;
    case 7:
      if (half_n) gemm_k<7,2><<<grid, blk, 0, stream>>>(A, Bt, bias, out, Mrows, Ncols, K, ldout, scale, sA, sB, sO);
      else        gemm_k<7,4><<<grid, blk, 0, stream>>>(A, Bt, bias, out, Mrows, Ncols, K, ldout, scale, sA, sB, sO);
      break;
    case 8: gemm_k<8,4><<<grid, blk, 0, stream>>>(A, Bt, bias, out, Mrows, Ncols, K, ldout, scale, sA, sB, sO); break;
    case 9: gemm_k<9,4><<<grid, blk, 0, stream>>>(A, Bt, bias, out, Mrows, Ncols, K, ldout, scale, sA, sB, sO); break;
    default: break;
  }
}

extern "C" void kernel_launch(void* const* d_in, const int* in_sizes, int n_in,
                              void* d_out, int out_size, void* d_ws, size_t ws_size,
                              hipStream_t stream)
{
  (void)in_sizes; (void)n_in; (void)out_size;
  // --- inputs (jax pytree flatten: dicts in sorted-key order) ---
  const float* hs      = (const float*)d_in[0];   // (B,R,S,D)
  const float* amask   = (const float*)d_in[1];   // (B,S,S)
  const float* dist    = (const float*)d_in[2];   // (B,S,S)
  const float* final_b = (const float*)d_in[3];
  const float* final_g = (const float*)d_in[4];
  const float* c_kb = (const float*)d_in[5];  const float* c_kw = (const float*)d_in[6];
  const float* c_ob = (const float*)d_in[7];  const float* c_ow = (const float*)d_in[8];
  const float* c_qb = (const float*)d_in[9];  const float* c_qw = (const float*)d_in[10];
  const float* c_vb = (const float*)d_in[11]; const float* c_vw = (const float*)d_in[12];
  const float* f1b  = (const float*)d_in[13]; const float* f1w  = (const float*)d_in[14];
  const float* f2b  = (const float*)d_in[15]; const float* f2w  = (const float*)d_in[16];
  const float* ln_c_b = (const float*)d_in[17]; const float* ln_c_g = (const float*)d_in[18];
  const float* ln_f_b = (const float*)d_in[19]; const float* ln_f_g = (const float*)d_in[20];
  const float* ln_r_b = (const float*)d_in[21]; const float* ln_r_g = (const float*)d_in[22];
  const float* bias_emb = (const float*)d_in[23]; // (L,NB,H)
  const float* r_kb = (const float*)d_in[24]; const float* r_kw = (const float*)d_in[25];
  const float* r_ob = (const float*)d_in[26]; const float* r_ow = (const float*)d_in[27];
  const float* r_qb = (const float*)d_in[28]; const float* r_qw = (const float*)d_in[29];
  const float* r_vb = (const float*)d_in[30]; const float* r_vw = (const float*)d_in[31];

  // --- workspace arena ---
  const size_t SZ_X   = (size_t)NT * DIM * 4;          //  67.1 MB residual f32
  const size_t SZ_H   = (size_t)NT * DIM * 2;          //  33.5 MB bf16 LN out
  const size_t PL     = 4194304;                       //  bf16 weight elems/layer
  const size_t SZ_WT  = (size_t)LNUM * PL * 2;         //  16.8 MB
  const size_t QKVELT = (size_t)NH * BB * SS * RR * HDIM;  // 16,777,216 elems
  const size_t SZ_ATT = 4 * QKVELT * 2;                // QP+KP+VT (2B) + SC (4B) = 134.2 MB
  const size_t SZ_PB  = (size_t)NH * SS * BB * RR * RR * 2; // 16.8 MB
  const size_t SZ_CT  = (size_t)NT * DIM * 2;          //  33.5 MB
  if (ws_size < SZ_X + SZ_H + SZ_WT + SZ_ATT + SZ_PB + SZ_CT) return;

  char* p = (char*)d_ws;
  float*  X  = (float*)p;  p += SZ_X;
  __bf16* Hb = (__bf16*)p; p += SZ_H;
  __bf16* WT = (__bf16*)p; p += SZ_WT;
  char* attn = p;          p += SZ_ATT;
  __bf16* QP = (__bf16*)attn;
  __bf16* KP = QP + QKVELT;
  __bf16* VT = KP + QKVELT;
  float*  SC = (float*)(VT + QKVELT);
  __bf16* F1 = (__bf16*)attn;          // FFN intermediate overlays QP..SC
  __bf16* PB = (__bf16*)p; p += SZ_PB;
  __bf16* CT = (__bf16*)p; p += SZ_CT;

  // --- residual init: X = hidden_states (already (B,R,S,D) token-major) ---
  hipMemcpyAsync(X, hs, SZ_X, hipMemcpyDeviceToDevice, stream);

  // --- convert + transpose all weights to bf16 (Wt[m][k] = W[k][m]) ---
  auto wconv = [&](const float* W, __bf16* Wt, int Kd, int Md) {
    long long tot = (long long)Kd * Md;
    wconv_k<<<(unsigned)((tot + 255) / 256), 256, 0, stream>>>(W, Wt, Kd, Md);
  };
  for (int l = 0; l < LNUM; ++l) {
    __bf16* base = WT + (size_t)l * PL;
    wconv(r_qw + (size_t)l*DIM*DIM,  base + 0,       DIM,  DIM);
    wconv(r_kw + (size_t)l*DIM*DIM,  base + 262144,  DIM,  DIM);
    wconv(r_vw + (size_t)l*DIM*DIM,  base + 524288,  DIM,  DIM);
    wconv(r_ow + (size_t)l*DIM*DIM,  base + 786432,  DIM,  DIM);
    wconv(c_qw + (size_t)l*DIM*DIM,  base + 1048576, DIM,  DIM);
    wconv(c_kw + (size_t)l*DIM*DIM,  base + 1310720, DIM,  DIM);
    wconv(c_vw + (size_t)l*DIM*DIM,  base + 1572864, DIM,  DIM);
    wconv(c_ow + (size_t)l*DIM*DIM,  base + 1835008, DIM,  DIM);
    wconv(f1w  + (size_t)l*DIM*IDIM, base + 2097152, DIM,  IDIM);
    wconv(f2w  + (size_t)l*IDIM*DIM, base + 3145728, IDIM, DIM);
  }

  const float rowq_scale = 0.022097086912079610f;   // (HD^-0.5)/sqrt(R)
  const float colq_scale = 0.125f;                  // HD^-0.5

  for (int l = 0; l < LNUM; ++l) {
    __bf16* Wrq = WT + (size_t)l*PL + 0;       __bf16* Wrk = WT + (size_t)l*PL + 262144;
    __bf16* Wrv = WT + (size_t)l*PL + 524288;  __bf16* Wro = WT + (size_t)l*PL + 786432;
    __bf16* Wcq = WT + (size_t)l*PL + 1048576; __bf16* Wck = WT + (size_t)l*PL + 1310720;
    __bf16* Wcv = WT + (size_t)l*PL + 1572864; __bf16* Wco = WT + (size_t)l*PL + 1835008;
    __bf16* Wf1 = WT + (size_t)l*PL + 2097152; __bf16* Wf2 = WT + (size_t)l*PL + 3145728;

    // ---------------- row self-attention ----------------
    ln_k<true><<<NT/8, 256, 0, stream>>>(X, ln_r_g + l*DIM, ln_r_b + l*DIM, Hb);
    launch_gemm(3, Hb, Wrq, r_qb + l*DIM, QP, NT, DIM, DIM, 0, rowq_scale, 0,0,0, 1, stream);
    launch_gemm(3, Hb, Wrk, r_kb + l*DIM, KP, NT, DIM, DIM, 0, 1.f,        0,0,0, 1, stream);
    launch_gemm(4, Hb, Wrv, r_vb + l*DIM, VT, NT, DIM, DIM, 0, 1.f,        0,0,0, 1, stream);
    // scores[z=(h,b)]: (S x S) = Q(S x 2048) x K^T, K-dim = R*HD
    launch_gemm(7, QP, KP, nullptr, SC, SS, SS, RR*HDIM, SS, 1.f,
                (long long)SS*RR*HDIM, (long long)SS*RR*HDIM, (long long)SS*SS,
                NH*BB, stream);
    softmax_row_k<<<(NH*BB*SS)/8, 256, 0, stream>>>(SC, amask, dist,
                                                    bias_emb + l*NBK*NH, PB);
    // ctx[z=(h,b)]: (S x R*HD) = P(S x S) x Vt^T
    launch_gemm(8, PB, VT, nullptr, CT, SS, RR*HDIM, SS, 0, 1.f,
                (long long)SS*SS, (long long)RR*HDIM*SS, 0, NH*BB, stream);
    launch_gemm(0, CT, Wro, r_ob + l*DIM, X, NT, DIM, DIM, DIM, 1.f, 0,0,0, 1, stream);

    // ---------------- column self-attention ----------------
    ln_k<true><<<NT/8, 256, 0, stream>>>(X, ln_c_g + l*DIM, ln_c_b + l*DIM, Hb);
    launch_gemm(5, Hb, Wcq, c_qb + l*DIM, QP, NT, DIM, DIM, 0, colq_scale, 0,0,0, 1, stream);
    launch_gemm(5, Hb, Wck, c_kb + l*DIM, KP, NT, DIM, DIM, 0, 1.f,        0,0,0, 1, stream);
    launch_gemm(6, Hb, Wcv, c_vb + l*DIM, VT, NT, DIM, DIM, 0, 1.f,        0,0,0, 1, stream);
    // scores[z=(h,s,b)]: (R x R), K-dim = HD  (N=32 -> NSUB=2 instantiation)
    launch_gemm(7, QP, KP, nullptr, SC, RR, RR, HDIM, RR, 1.f,
                (long long)RR*HDIM, (long long)RR*HDIM, (long long)RR*RR,
                NH*SS*BB, stream);
    softmax_col_k<<<(NH*SS*BB*RR)/8, 256, 0, stream>>>(SC, PB);
    // ctx[z=(h,s,b)]: (R x HD) = P(R x R) x Vt^T, K-dim = R
    launch_gemm(9, PB, VT, nullptr, CT, RR, HDIM, RR, 0, 1.f,
                (long long)RR*RR, (long long)HDIM*RR, 0, NH*SS*BB, stream);
    launch_gemm(0, CT, Wco, c_ob + l*DIM, X, NT, DIM, DIM, DIM, 1.f, 0,0,0, 1, stream);

    // ---------------- FFN ----------------
    ln_k<true><<<NT/8, 256, 0, stream>>>(X, ln_f_g + l*DIM, ln_f_b + l*DIM, Hb);
    launch_gemm(2, Hb, Wf1, f1b + l*IDIM, F1, NT, IDIM, DIM, IDIM, 1.f, 0,0,0, 1, stream);
    launch_gemm(0, F1, Wf2, f2b + l*DIM,  X,  NT, DIM, IDIM, DIM,  1.f, 0,0,0, 1, stream);
  }

  // final LayerNorm -> f32 output (B,R,S,D)
  ln_k<false><<<NT/8, 256, 0, stream>>>(X, final_g, final_b, d_out);
}
